// AttentionNN_53171695125358
// MI455X (gfx1250) — matrix operand order repaired and checked by reading, not compiled, not run on hardware
//
#include <hip/hip_runtime.h>

// Problem sizes (fixed by the reference)
#define NROWS  65536
#define DIN    64
#define NCTRS  1024
#define DOUT   256
#define KB     32                 // centers per k-block
#define NB     (NCTRS / KB)       // 32 blocks
#define LOG2E      1.4426950408889634f
#define TWO_LOG2E  2.8853900817779268f
#define OVERSHOOT  12.0f          // log2-domain headroom to make rescales rare

typedef __attribute__((ext_vector_type(16))) _Float16 v16h;
typedef __attribute__((ext_vector_type(8)))  float    v8f;
typedef __attribute__((ext_vector_type(4)))  float    v4f;
typedef __attribute__((ext_vector_type(4)))  unsigned v4u;
typedef __attribute__((ext_vector_type(8)))  int      v8i;
typedef __attribute__((ext_vector_type(4)))  int      v4i;

// Raw v_exp_f32: args are <= 0; flush-to-zero on deep underflow is exactly
// what softmax tails want, so skip libm's ldexp fixup path.
__device__ __forceinline__ float fast_exp2(float a) {
#if __has_builtin(__builtin_amdgcn_exp2f)
    return __builtin_amdgcn_exp2f(a);
#else
    return exp2f(a);
#endif
}

// ---------------------------------------------------------------------------
// Prep: fold s into centers, emit f16 centers in natural [k][d] order,
// c_sq pre-scaled by log2(e), and values transposed to f16 vT[d][k].
// (x_sq is NOT needed: it shifts all logits of a row equally -> cancels in softmax.)
// ---------------------------------------------------------------------------
__global__ __launch_bounds__(256) void prep_params(
    const float* __restrict__ ctrs, const float* __restrict__ values,
    const float* __restrict__ s,
    _Float16* __restrict__ ch, _Float16* __restrict__ vT,
    float* __restrict__ csq2)
{
    int k = blockIdx.x * 256 + threadIdx.x;
    if (k >= NCTRS) return;
    float acc = 0.0f;
    #pragma unroll 8
    for (int d = 0; d < DIN; ++d) {
        float c  = ctrs[k * DIN + d];
        float sv = s[d];
        acc += c * c * sv;
        // keep natural order within each 32-dim block so each A fragment is one
        // contiguous 32B run per lane: lane half h holds K=16h..16h+15
        int b = d >> 5, sub = d & 31, grp = sub >> 3, e = sub & 7;
        int gp = grp;   // identity
        ch[k * DIN + b * 32 + gp * 8 + e] = (_Float16)(c * sv);
    }
    csq2[k] = acc * LOG2E;
    #pragma unroll 8
    for (int d = 0; d < DOUT; ++d)
        vT[(size_t)d * NCTRS + k] = (_Float16)values[k * DOUT + d];
}

// ---------------------------------------------------------------------------
// Tensor Data Mover: 2-D tile (2-byte elements) global -> LDS.
// D# group0/group1 per CDNA5 ISA §8; groups 2/3 zero (<=2D tensor).
// ---------------------------------------------------------------------------
__device__ __forceinline__ void tdm_load_2d(
    unsigned lds_addr, const void* gptr,
    unsigned tile_d0, unsigned tile_d1,
    unsigned tensor_d0, unsigned tensor_d1,
    unsigned long long stride0)
{
    unsigned long long ga = (unsigned long long)(uintptr_t)gptr;
    v4u g0;
    g0[0] = 1u;                                        // count=1, user mode
    g0[1] = lds_addr;                                  // LDS byte address
    g0[2] = (unsigned)ga;                              // global addr lo
    g0[3] = (unsigned)((ga >> 32) & 0x01FFFFFFu) | (2u << 30);  // hi + type=2
    v8i g1;
    g1[0] = (int)(1u << 16);                           // data_size=1 (2B), mask=0
    g1[1] = (int)((tensor_d0 & 0xFFFFu) << 16);        // tensor_dim0[15:0]
    g1[2] = (int)(((tensor_d0 >> 16) & 0xFFFFu) |
                  ((tensor_d1 & 0xFFFFu) << 16));      // tensor_dim0[31:16] | tensor_dim1[15:0]
    g1[3] = (int)(((tensor_d1 >> 16) & 0xFFFFu) |
                  (tile_d0 << 16));                    // tensor_dim1[31:16] | tile_dim0
    g1[4] = (int)(tile_d1 & 0xFFFFu);                  // tile_dim1 (tile_dim2=0)
    g1[5] = (int)(unsigned)stride0;                    // tensor_dim0_stride lo32
    g1[6] = (int)((stride0 >> 32) & 0xFFFFu);          // stride hi16 (dim1_stride=0)
    g1[7] = 0;
    v4i z4 = {0, 0, 0, 0};
    v8i z8 = {0, 0, 0, 0, 0, 0, 0, 0};
    __builtin_amdgcn_tensor_load_to_lds(g0, g1, z4, z4, z8, 0);
}

// ---------------------------------------------------------------------------
// Fused kernel: 256 threads (8 wave32) x 128 rows; flash-style online softmax
// over 1024 centers in blocks of 32, TDM-staged double-buffered LDS tiles.
// ---------------------------------------------------------------------------
__global__ __launch_bounds__(256) void fused_rbf_attn(
    const float*    __restrict__ x,
    const _Float16* __restrict__ ch,    // [NCTRS][DIN] f16, natural order, s folded
    const _Float16* __restrict__ vT,    // [DOUT][NCTRS] f16
    const float*    __restrict__ csq2,  // [NCTRS], pre-scaled by log2(e)
    float*          __restrict__ out)   // [NROWS][DOUT]
{
    __shared__ __attribute__((aligned(16))) _Float16 lds_c[2][KB * DIN];    // 2 x 4 KB
    __shared__ __attribute__((aligned(16))) _Float16 lds_v[2][DOUT * KB];   // 2 x 16 KB
    __shared__ __attribute__((aligned(32))) float    lds_csq[NCTRS];        // 4 KB

    const int tid  = threadIdx.x;
    const int wave = tid >> 5;
    const int lane = tid & 31;
    const int half = lane >> 4;
    const int ln   = lane & 15;
    const int wrow = blockIdx.x * 128 + wave * 16;     // this wave's 16 rows

    // stage pre-scaled c_sq once (1024 f32)
    ((v4f*)lds_csq)[tid] = ((const v4f*)csq2)[tid];

    // ---- x-side B fragments (B 32x16: lane<16 K=0..15, lane>=16 K=16..31) ----
    v16h bx[2];
    {
        const float* xr = x + (size_t)(wrow + ln) * DIN;
        #pragma unroll
        for (int b = 0; b < 2; ++b) {
            const int d0 = half * 16 + 32 * b;
            #pragma unroll
            for (int q = 0; q < 4; ++q) {
                v4f f = *(const v4f*)(xr + d0 + 4 * q);
                bx[b][4 * q + 0] = (_Float16)f[0];
                bx[b][4 * q + 1] = (_Float16)f[1];
                bx[b][4 * q + 2] = (_Float16)f[2];
                bx[b][4 * q + 3] = (_Float16)f[3];
            }
        }
    }

    // ---- accumulators / softmax state ----
    v8f O[16];
    #pragma unroll
    for (int t = 0; t < 16; ++t)
        #pragma unroll
        for (int r = 0; r < 8; ++r) O[t][r] = 0.0f;
    float Mx = -__builtin_inff();
    float Ls = 0.0f;

    // ---- TDM prologue: stage block 0 into buffer 0 ----
    if (wave == 0) {
        tdm_load_2d((unsigned)(uintptr_t)&lds_c[0][0], ch,
                    KB * DIN, 1, NCTRS * DIN, 1, KB * DIN);
        tdm_load_2d((unsigned)(uintptr_t)&lds_v[0][0], vT,
                    KB, DOUT, NCTRS, DOUT, NCTRS);
    }

    for (int kb = 0; kb < NB; ++kb) {
        const int buf = kb & 1;
        if (wave == 0) __builtin_amdgcn_s_wait_tensorcnt(0);
        __syncthreads();   // tile kb visible to all; prev reads of buf^1 done
        if (wave == 0 && kb + 1 < NB) {
            tdm_load_2d((unsigned)(uintptr_t)&lds_c[buf ^ 1][0],
                        ch + (size_t)(kb + 1) * KB * DIN,
                        KB * DIN, 1, NCTRS * DIN, 1, KB * DIN);
            tdm_load_2d((unsigned)(uintptr_t)&lds_v[buf ^ 1][0],
                        vT + (size_t)(kb + 1) * KB,
                        KB, DOUT, NCTRS, DOUT, NCTRS);
        }

        // ---- cross GEMM (transposed): A = centers (M), B = x rows (N) ----
        // natural ch layout => each fragment is one contiguous 32B LDS read
        const _Float16* lc = &lds_c[buf][0];
        v8f S0, S1;
        #pragma unroll
        for (int r = 0; r < 8; ++r) { S0[r] = 0.0f; S1[r] = 0.0f; }
        {
            v16h a00 = *(const v16h*)&lc[(0 * 16 + ln) * DIN +  0 + half * 16];
            v16h a01 = *(const v16h*)&lc[(0 * 16 + ln) * DIN + 32 + half * 16];
            S0 = __builtin_amdgcn_wmma_f32_16x16x32_f16(false, a00, false, bx[0], (short)0, S0, false, false);
            S0 = __builtin_amdgcn_wmma_f32_16x16x32_f16(false, a01, false, bx[1], (short)0, S0, false, false);
            v16h a10 = *(const v16h*)&lc[(1 * 16 + ln) * DIN +  0 + half * 16];
            v16h a11 = *(const v16h*)&lc[(1 * 16 + ln) * DIN + 32 + half * 16];
            S1 = __builtin_amdgcn_wmma_f32_16x16x32_f16(false, a10, false, bx[0], (short)0, S1, false, false);
            S1 = __builtin_amdgcn_wmma_f32_16x16x32_f16(false, a11, false, bx[1], (short)0, S1, false, false);
        }

        // ---- logits (base-2), x_sq dropped (row-constant cancels in softmax) ----
        const int kbase = kb * KB;
        v8f c0 = *(const v8f*)&lds_csq[kbase + 8 * half];
        v8f c1 = *(const v8f*)&lds_csq[kbase + 16 + 8 * half];
        #pragma unroll
        for (int r = 0; r < 8; ++r) {
            S0[r] = S0[r] * TWO_LOG2E - c0[r];   // single v_pk_fma pairwise
            S1[r] = S1[r] * TWO_LOG2E - c1[r];
        }

        // ---- block max; rescale only when a new record appears (rare) ----
        float bm = fmaxf(S0[0], S0[1]);
        #pragma unroll
        for (int r = 2; r < 8; ++r) bm = fmaxf(bm, S0[r]);
        #pragma unroll
        for (int r = 0; r < 8; ++r) bm = fmaxf(bm, S1[r]);
        bm = fmaxf(bm, __shfl_xor(bm, 16));
        const bool need = bm > Mx;
        if (__any(need)) {
            const float Mn    = need ? bm + OVERSHOOT : Mx;  // overshoot: few future records
            const float alpha = fast_exp2(Mx - Mn);          // 0 on first block
            float av[8];
            #pragma unroll
            for (int r = 0; r < 8; ++r) av[r] = __shfl(alpha, r + 8 * half);
            #pragma unroll
            for (int t = 0; t < 16; ++t)
                #pragma unroll
                for (int r = 0; r < 8; ++r) O[t][r] *= av[r];
            Ls *= alpha;
            Mx  = Mn;
        }

        // ---- exp weights; they land exactly in A-fragment layout per lane ----
        float e[16];
        #pragma unroll
        for (int r = 0; r < 8; ++r) { S0[r] = fast_exp2(S0[r] - Mx); e[r]     = S0[r]; }
        #pragma unroll
        for (int r = 0; r < 8; ++r) { S1[r] = fast_exp2(S1[r] - Mx); e[8 + r] = S1[r]; }
        // pairwise tree sum (short dependency chain)
        #pragma unroll
        for (int st = 1; st < 16; st <<= 1)
            #pragma unroll
            for (int i = 0; i < 16; i += 2 * st) e[i] += e[i + st];
        float ps = e[0] + __shfl_xor(e[0], 16);
        Ls += ps;

        v16h P;
        #pragma unroll
        for (int q = 0; q < 8; ++q) {
            P[q]     = (_Float16)S0[q];
            P[8 + q] = (_Float16)S1[q];
        }

        // ---- second GEMM: O[n][d] += P(16x32) @ V(32x16) per d-tile ----
        const _Float16* lv = &lds_v[buf][0];
        #pragma unroll
        for (int t = 0; t < 16; ++t) {
            const _Float16* bp = &lv[(16 * t + ln) * KB + 8 * half];  // table order: K=8h..8h+7, 16+8h..16+8h+7
            v16h Bv;
            #pragma unroll
            for (int s = 0; s < 8; ++s) { Bv[s] = bp[s]; Bv[8 + s] = bp[16 + s]; }
            O[t] = __builtin_amdgcn_wmma_f32_16x16x32_f16(
                       false, P, false, Bv, (short)0, O[t], false, false);
        }
    }

    // ---- finalize: divide by denominator, store ----
    const float Linv = 1.0f / Ls;
    float lv8[8];
    #pragma unroll
    for (int r = 0; r < 8; ++r) lv8[r] = __shfl(Linv, r + 8 * half);
    #pragma unroll
    for (int t = 0; t < 16; ++t) {
        #pragma unroll
        for (int r = 0; r < 8; ++r) {
            const int n = wrow + r + 8 * half;
            out[(size_t)n * DOUT + 16 * t + ln] = O[t][r] * lv8[r];
        }
    }
}

// ---------------------------------------------------------------------------
extern "C" void kernel_launch(void* const* d_in, const int* in_sizes, int n_in,
                              void* d_out, int out_size, void* d_ws, size_t ws_size,
                              hipStream_t stream)
{
    const float* x      = (const float*)d_in[0];
    const float* ctrs   = (const float*)d_in[1];
    const float* values = (const float*)d_in[2];
    const float* s      = (const float*)d_in[3];
    float* out = (float*)d_out;

    char* ws = (char*)d_ws;
    const size_t CH_B = (size_t)NCTRS * DIN * sizeof(_Float16);    // 128 KB
    const size_t VT_B = (size_t)DOUT * NCTRS * sizeof(_Float16);   // 512 KB
    _Float16* ch   = (_Float16*)(ws);
    _Float16* vT   = (_Float16*)(ws + CH_B);
    float*    csq2 = (float*)(ws + CH_B + VT_B);

    prep_params<<<NCTRS / 256, 256, 0, stream>>>(ctrs, values, s, ch, vT, csq2);
    fused_rbf_attn<<<NROWS / 128, 256, 0, stream>>>(x, ch, vT, csq2, out);
}
